// LLM_MAIN_37391985279375
// MI455X (gfx1250) — matrix operand order repaired and checked by reading, not compiled, not run on hardware
//
#include <hip/hip_runtime.h>
#include <hip/hip_bf16.h>

// ---------------- model dims ----------------
constexpr int LAYERS = 2;
constexpr int SEQ    = 1024;
constexpr int HID    = 2048;
constexpr int NHEAD  = 16;
constexpr int KVH    = 8;
constexpr int HD     = 128;   // head dim
constexpr int QKH    = NHEAD + KVH;        // 24 (q+k heads, get rmsnorm+rope)
constexpr int NQKVH  = QKH + KVH;          // 32 total heads in qkv proj
constexpr int NQKV   = NQKVH * HD;         // 4096
constexpr int DFF    = 6144;
constexpr int VOCAB  = 32000;
constexpr int GQA    = NHEAD / KVH;        // 2

// ---------------- vector types ----------------
typedef __attribute__((ext_vector_type(16))) _Float16 v16h;
typedef __attribute__((ext_vector_type(8)))  _Float16 h8;
typedef __attribute__((ext_vector_type(8)))  float    v8f;
typedef __attribute__((ext_vector_type(16))) float    f16v;
typedef __attribute__((ext_vector_type(4)))  int      v4i;

// ---------------- async to-LDS support (gfx1250) ----------------
#if defined(__has_builtin)
#if __has_builtin(__builtin_amdgcn_global_load_async_to_lds_b128)
#define USE_ASYNC_LDS 1
#endif
#endif

static __device__ inline void stage_b128(const float* g, float* l) {
#if defined(USE_ASYNC_LDS)
  __builtin_amdgcn_global_load_async_to_lds_b128(
      (__attribute__((address_space(1))) v4i*)g,
      (__attribute__((address_space(3))) v4i*)l, 0, 0);
#else
  *(float4*)l = *(const float4*)g;
#endif
}

static __device__ inline void wait_async_all() {
#if defined(USE_ASYNC_LDS)
#if __has_builtin(__builtin_amdgcn_s_wait_asynccnt)
  __builtin_amdgcn_s_wait_asynccnt(0);
#else
  asm volatile("s_wait_asynccnt 0x0" ::: "memory");
#endif
#endif
}

// ---------------- helpers ----------------
static __device__ inline v8f vzero8() {
  v8f r;
#pragma unroll
  for (int i = 0; i < 8; ++i) r[i] = 0.0f;
  return r;
}
// pack two 8-float chunks (A layout halves) into one f16 A fragment
static __device__ inline v16h cvt_pair_f32(const float* lo, const float* hi) {
  v16h r;
#pragma unroll
  for (int i = 0; i < 8; ++i) { r[i] = (_Float16)lo[i]; r[i + 8] = (_Float16)hi[i]; }
  return r;
}
static __device__ inline v16h pack_h8(h8 lo, h8 hi) {
  v16h r;
#pragma unroll
  for (int i = 0; i < 8; ++i) { r[i] = lo[i]; r[i + 8] = hi[i]; }
  return r;
}

#define WMMA_F16(A, B, C) \
  __builtin_amdgcn_wmma_f32_16x16x32_f16(false, (A), false, (B), (short)0, (C), false, false)

// ---------------- rmsnorm: out = x * rsqrt(sum(x^2)) ----------------
__global__ __launch_bounds__(256) void k_rmsnorm(const float* __restrict__ in,
                                                 float* __restrict__ out, int dim) {
  const int row = blockIdx.x;
  const float* x = in + (size_t)row * dim;
  float ss = 0.0f;
  for (int i = threadIdx.x; i < dim; i += blockDim.x) { float v = x[i]; ss += v * v; }
#pragma unroll
  for (int m = 16; m > 0; m >>= 1) ss += __shfl_xor(ss, m, 32);
  __shared__ float red[8];
  const int wid = threadIdx.x >> 5, lane = threadIdx.x & 31;
  if (lane == 0) red[wid] = ss;
  __syncthreads();
  if (threadIdx.x == 0) {
    float t = 0.0f;
    const int nw = blockDim.x >> 5;
    for (int i = 0; i < nw; ++i) t += red[i];
    red[0] = rsqrtf(t);
  }
  __syncthreads();
  const float r = red[0];
  for (int i = threadIdx.x; i < dim; i += blockDim.x)
    out[(size_t)row * dim + i] = x[i] * r;
}

// ---------------- qk rmsnorm + qk_norm_w + rope; v -> f16 transposed ----------------
__global__ __launch_bounds__(128) void k_qkrope(const float* __restrict__ qkv,
                                                const float* __restrict__ nw,
                                                const float* __restrict__ cosp,
                                                const float* __restrict__ sinp,
                                                _Float16* __restrict__ q16,
                                                _Float16* __restrict__ k16,
                                                _Float16* __restrict__ vt16) {
  const int s = blockIdx.x, head = blockIdx.y, d = threadIdx.x;
  const float* x = qkv + ((size_t)s * NQKVH + head) * HD;
  if (head >= QKH) {  // v head: plain f16 cast, store transposed
    const int kh = head - QKH;
    vt16[((size_t)kh * HD + d) * SEQ + s] = (_Float16)x[d];
    return;
  }
  const float v = x[d];
  float ss = v * v;
#pragma unroll
  for (int m = 16; m > 0; m >>= 1) ss += __shfl_xor(ss, m, 32);
  __shared__ float red[4];
  if ((d & 31) == 0) red[d >> 5] = ss;
  __syncthreads();
  const float r = rsqrtf(red[0] + red[1] + red[2] + red[3]);
  __shared__ float y[HD];
  y[d] = v * r * nw[(size_t)head * HD + d];
  __syncthreads();
  // reference rotate_half: rh[i] = x[(i+64) % 128] (flip of the split, no negation)
  const float out = y[d] * cosp[(size_t)s * HD + d] + y[(d + 64) & (HD - 1)] * sinp[(size_t)s * HD + d];
  if (head < NHEAD) q16[((size_t)head * SEQ + s) * HD + d] = (_Float16)out;
  else              k16[((size_t)(head - NHEAD) * SEQ + s) * HD + d] = (_Float16)out;
}

// ---------------- WMMA GEMM: C[M][N] = A[M][K] * W[N][K]^T (+res) ----------------
// grid = (N/128, M/128), block = 256 (8 waves, 4x2), wave tile = 32x64.
// A tile (f32) staged via async-to-LDS; W tile staged as f16 (converted once).
// Double-buffered LDS, one barrier per K-step.
__global__ __launch_bounds__(256) void k_gemm_wmma(float* __restrict__ C,
                                                   const float* __restrict__ A,
                                                   const float* __restrict__ W,
                                                   const float* __restrict__ res,
                                                   int M, int N, int K) {
  __shared__ __align__(64) float    Asm[2][128][32];  // 32 KB
  __shared__ __align__(64) _Float16 Bsm[2][128][32];  // 16 KB

  const int tid  = threadIdx.x;
  const int lane = tid & 31;
  const int wave = tid >> 5;
  const int wm = wave & 3;        // 4 M groups of 32
  const int wn = wave >> 2;       // 2 N groups of 64
  const int mBase = blockIdx.y * 128;
  const int nBase = blockIdx.x * 128;
  const int m0loc = wm * 32;
  const int n0loc = wn * 64;
  const int rA = lane & 15;
  const int hiHalf = (lane >> 4) & 1;
  const int ka = hiHalf ? 16 : 0;   // A frag K offsets 0..15 / 16..31
  const int kb = hiHalf ? 16 : 0;   // B frag K offsets 0..15 / 16..31

  // staging decomposition: 256 threads cover 128 rows x 2 halves of 16 elems
  const int srow = tid >> 1;
  const int scol = (tid & 1) * 16;

  v8f acc[2][4];
#pragma unroll
  for (int i = 0; i < 2; ++i)
#pragma unroll
    for (int j = 0; j < 4; ++j) acc[i][j] = vzero8();

  // prologue: stage k=0 into buffer 0
  {
    const float* ga = A + (size_t)(mBase + srow) * K + scol;
    float* la = &Asm[0][srow][scol];
#pragma unroll
    for (int i = 0; i < 4; ++i) stage_b128(ga + i * 4, la + i * 4);
    const float* gb = W + (size_t)(nBase + srow) * K + scol;
    v16h hv;
#pragma unroll
    for (int i = 0; i < 16; ++i) hv[i] = (_Float16)gb[i];
    *(v16h*)&Bsm[0][srow][scol] = hv;
  }

  int buf = 0;
  for (int k = 0; k < K; k += 32) {
    wait_async_all();
    __syncthreads();

    if (k + 32 < K) {  // stage next chunk into other buffer
      const float* ga = A + (size_t)(mBase + srow) * K + (k + 32) + scol;
      float* la = &Asm[buf ^ 1][srow][scol];
#pragma unroll
      for (int i = 0; i < 4; ++i) stage_b128(ga + i * 4, la + i * 4);
      const float* gb = W + (size_t)(nBase + srow) * K + (k + 32) + scol;
      v16h hv;
#pragma unroll
      for (int i = 0; i < 16; ++i) hv[i] = (_Float16)gb[i];
      *(v16h*)&Bsm[buf ^ 1][srow][scol] = hv;
    }

    // fragments from LDS
    v16h fa[2], fb[4];
#pragma unroll
    for (int i = 0; i < 2; ++i) {
      const float* ap = &Asm[buf][m0loc + i * 16 + rA][0];
      fa[i] = cvt_pair_f32(ap + ka, ap + 8 + ka);
    }
#pragma unroll
    for (int j = 0; j < 4; ++j)
      fb[j] = *(const v16h*)&Bsm[buf][n0loc + j * 16 + rA][kb];

#pragma unroll
    for (int i = 0; i < 2; ++i)
#pragma unroll
      for (int j = 0; j < 4; ++j)
        acc[i][j] = WMMA_F16(fa[i], fb[j], acc[i][j]);

    buf ^= 1;
  }

  // C/D layout: VGPR r, lane -> row = r + (lane<16?0:8), col = lane&15
#pragma unroll
  for (int i = 0; i < 2; ++i) {
#pragma unroll
    for (int j = 0; j < 4; ++j) {
#pragma unroll
      for (int r = 0; r < 8; ++r) {
        const int row = mBase + m0loc + i * 16 + r + hiHalf * 8;
        const int col = nBase + n0loc + j * 16 + rA;
        float v = acc[i][j][r];
        if (res) v += res[(size_t)row * N + col];
        C[(size_t)row * N + col] = v;
      }
    }
  }
}

// ---------------- flash attention (one wave per 16 query rows x head) ----------------
// q:[NHEAD][SEQ][HD] f16, k:[KVH][SEQ][HD] f16, vt:[KVH][HD][SEQ] f16, o:[SEQ][HID] f32
__global__ __launch_bounds__(32) void k_attn(const _Float16* __restrict__ q,
                                             const _Float16* __restrict__ kk,
                                             const _Float16* __restrict__ vt,
                                             float* __restrict__ o) {
  const int lane = threadIdx.x & 31;
  const int qb = blockIdx.x;      // query block of 16 rows
  const int h  = blockIdx.y;      // q head
  const int kh = h / GQA;         // kv head
  const int rA = lane & 15;
  const int hiHalf = (lane >> 4) & 1;
  const int ka = hiHalf ? 16 : 0;
  const int kb = hiHalf ? 16 : 0;

  // load Q fragments once: 16x128 = 4 A-fragments of 16x32
  const _Float16* qp = q + ((size_t)h * SEQ + (qb * 16 + rA)) * HD;
  v16h qf[4];
#pragma unroll
  for (int c = 0; c < 4; ++c) {
    h8 lo = *(const h8*)(qp + c * 32 + ka);
    h8 hi = *(const h8*)(qp + c * 32 + 8 + ka);
    qf[c] = pack_h8(lo, hi);
  }

  v8f oacc[8];
#pragma unroll
  for (int nd = 0; nd < 8; ++nd) oacc[nd] = vzero8();
  float mrow[8], lrow[8];
#pragma unroll
  for (int r = 0; r < 8; ++r) { mrow[r] = -1.0e30f; lrow[r] = 0.0f; }

  __shared__ __align__(64) _Float16 plds[16][32];

  for (int kt = 0; kt < SEQ; kt += 32) {
    // S = Q * K^T for 32 keys: two 16x16 score fragments
    v8f s0 = vzero8(), s1 = vzero8();
    const _Float16* kp0 = kk + ((size_t)kh * SEQ + kt + rA) * HD;
    const _Float16* kp1 = kp0 + 16 * HD;
#pragma unroll
    for (int c = 0; c < 4; ++c) {
      v16h b0 = *(const v16h*)(kp0 + c * 32 + kb);
      v16h b1 = *(const v16h*)(kp1 + c * 32 + kb);
      s0 = WMMA_F16(qf[c], b0, s0);
      s1 = WMMA_F16(qf[c], b1, s1);
    }

    // additive causal mask (-128, as in reference) + online softmax
#pragma unroll
    for (int r = 0; r < 8; ++r) {
      const int row = qb * 16 + r + hiHalf * 8;
      const int c0 = kt + rA, c1 = c0 + 16;
      float x0 = s0[r] + ((c0 > row) ? -128.0f : 0.0f);
      float x1 = s1[r] + ((c1 > row) ? -128.0f : 0.0f);
      float t = fmaxf(x0, x1);
#pragma unroll
      for (int m = 8; m > 0; m >>= 1) t = fmaxf(t, __shfl_xor(t, m, 32));
      const float nm = fmaxf(mrow[r], t);
      const float sc = __expf(mrow[r] - nm);
      const float p0 = __expf(x0 - nm);
      const float p1 = __expf(x1 - nm);
      float ps = p0 + p1;
#pragma unroll
      for (int m = 8; m > 0; m >>= 1) ps += __shfl_xor(ps, m, 32);
      lrow[r] = lrow[r] * sc + ps;
      mrow[r] = nm;
#pragma unroll
      for (int nd = 0; nd < 8; ++nd) oacc[nd][r] *= sc;
      plds[r + hiHalf * 8][rA]      = (_Float16)p0;
      plds[r + hiHalf * 8][rA + 16] = (_Float16)p1;
    }
    __syncthreads();

    // re-shape P from C-layout to A-fragment layout via LDS
    const _Float16* pr = &plds[rA][0];
    h8 plo = *(const h8*)(pr + ka);
    h8 phi = *(const h8*)(pr + 8 + ka);
    v16h pf = pack_h8(plo, phi);

    // O += P * V  (V transposed: contiguous keys per d-column)
#pragma unroll
    for (int nd = 0; nd < 8; ++nd) {
      const _Float16* vp = vt + ((size_t)kh * HD + nd * 16 + rA) * SEQ + kt + kb;
      v16h bv = *(const v16h*)vp;
      oacc[nd] = WMMA_F16(pf, bv, oacc[nd]);
    }
    __syncthreads();
  }

#pragma unroll
  for (int nd = 0; nd < 8; ++nd) {
#pragma unroll
    for (int r = 0; r < 8; ++r) {
      const int row = qb * 16 + r + hiHalf * 8;
      o[(size_t)row * HID + h * HD + nd * 16 + rA] = oacc[nd][r] / lrow[r];
    }
  }
}

// ---------------- silu(g) * u ----------------
__global__ __launch_bounds__(256) void k_silu_mul(const float* __restrict__ gu,
                                                  float* __restrict__ ff) {
  const int j = blockIdx.x * 256 + threadIdx.x;
  const int s = blockIdx.y;
  const float g = gu[(size_t)s * (2 * DFF) + j];
  const float u = gu[(size_t)s * (2 * DFF) + DFF + j];
  ff[(size_t)s * DFF + j] = (g / (1.0f + __expf(-g))) * u;
}

// ---------------- orchestration ----------------
extern "C" void kernel_launch(void* const* d_in, const int* in_sizes, int n_in,
                              void* d_out, int out_size, void* d_ws, size_t ws_size,
                              hipStream_t stream) {
  (void)in_sizes; (void)n_in; (void)out_size; (void)ws_size;
  const float* hs        = (const float*)d_in[0];
  const float* cosp      = (const float*)d_in[1];
  const float* sinp      = (const float*)d_in[2];
  const float* qkv_w     = (const float*)d_in[3];
  const float* qk_norm_w = (const float*)d_in[4];
  const float* o_w       = (const float*)d_in[5];
  const float* gate_up_w = (const float*)d_in[6];
  const float* down_w    = (const float*)d_in[7];
  const float* lm_head_w = (const float*)d_in[8];
  float* logits = (float*)d_out;

  float* h    = (float*)d_ws;
  float* xn   = h    + (size_t)SEQ * HID;
  float* qkvb = xn   + (size_t)SEQ * HID;
  float* obuf = qkvb + (size_t)SEQ * NQKV;
  float* gu   = obuf + (size_t)SEQ * HID;
  float* ff   = gu   + (size_t)SEQ * 2 * DFF;
  _Float16* q16  = (_Float16*)(ff + (size_t)SEQ * DFF);
  _Float16* k16  = q16 + (size_t)NHEAD * SEQ * HD;
  _Float16* vt16 = k16 + (size_t)KVH * SEQ * HD;

  (void)hipMemcpyAsync(h, hs, (size_t)SEQ * HID * sizeof(float),
                       hipMemcpyDeviceToDevice, stream);

  for (int l = 0; l < LAYERS; ++l) {
    k_rmsnorm<<<SEQ, 256, 0, stream>>>(h, xn, HID);
    k_gemm_wmma<<<dim3(NQKV / 128, SEQ / 128), 256, 0, stream>>>(
        qkvb, xn, qkv_w + (size_t)l * NQKV * HID, nullptr, SEQ, NQKV, HID);
    k_qkrope<<<dim3(SEQ, NQKVH), 128, 0, stream>>>(
        qkvb, qk_norm_w + (size_t)l * QKH * HD, cosp, sinp, q16, k16, vt16);
    k_attn<<<dim3(SEQ / 16, NHEAD), 32, 0, stream>>>(q16, k16, vt16, obuf);
    k_gemm_wmma<<<dim3(HID / 128, SEQ / 128), 256, 0, stream>>>(
        h, obuf, o_w + (size_t)l * HID * (NHEAD * HD), h, SEQ, HID, HID);
    k_rmsnorm<<<SEQ, 256, 0, stream>>>(h, xn, HID);
    k_gemm_wmma<<<dim3(2 * DFF / 128, SEQ / 128), 256, 0, stream>>>(
        gu, xn, gate_up_w + (size_t)l * 2 * DFF * HID, nullptr, SEQ, 2 * DFF, HID);
    k_silu_mul<<<dim3(DFF / 256, SEQ), 256, 0, stream>>>(gu, ff);
    k_gemm_wmma<<<dim3(HID / 128, SEQ / 128), 256, 0, stream>>>(
        h, ff, down_w + (size_t)l * HID * DFF, h, SEQ, HID, DFF);
  }
  k_rmsnorm<<<SEQ, 256, 0, stream>>>(h, xn, HID);
  k_gemm_wmma<<<dim3(VOCAB / 128, SEQ / 128), 256, 0, stream>>>(
      logits, xn, lm_head_w, nullptr, SEQ, VOCAB, HID);
}
